// OmniAnomaly_41832981463434
// MI455X (gfx1250) — compile-verified, matrix operand order checked by reading, not run on hardware
//
#include <hip/hip_runtime.h>
#include <hip/hip_bf16.h>

// ---------------- problem constants ----------------
constexpr int B  = 1024;
constexpr int T  = 100;
constexpr int XD = 38;
constexpr int H  = 500;
constexpr int ZD = 16;
constexpr int L  = 3;

constexpr int G3  = 3 * H;   // 1500
constexpr int G3P = 1504;    // 3H padded to /16
constexpr int KX  = 64;      // XD(38) padded to /32
constexpr int KH  = 512;     // H(500) padded to /32
constexpr int KC  = 544;     // H+ZD(516) padded to /32
constexpr int KZ  = 32;      // ZD(16) padded to /32
constexpr int NQ  = 32;      // [Wqm;Wqs] rows
constexpr int NP  = 80;      // [Wpm;Wps] rows (76 -> 80)

typedef __bf16 bf16_t;
typedef __attribute__((ext_vector_type(16))) __bf16 v16bf;
typedef __attribute__((ext_vector_type(8)))  __bf16 v8bf;
typedef __attribute__((ext_vector_type(8)))  float  v8f;

// ---------------- device helpers ----------------
__device__ __forceinline__ float softplusf(float x) {
  return (x > 20.0f) ? x : log1pf(__expf(x));
}
__device__ __forceinline__ float sigmoidf(float x) {
  return 1.0f / (1.0f + __expf(-x));
}

// Load one 16x32 bf16 A/B fragment row-slice for this lane.
// Per CDNA5 ISA layout: lane (r = lane&15, half = lane>>4) holds, for row r,
// K = [half*8 .. half*8+7] in vector elems 0..7 and
// K = [16+half*8 .. 16+half*8+7] in vector elems 8..15.
// p must point at row_base + half*8 (16B aligned).
__device__ __forceinline__ v16bf load_frag(const bf16_t* __restrict__ p) {
  v8bf lo = *reinterpret_cast<const v8bf*>(p);
  v8bf hi = *reinterpret_cast<const v8bf*>(p + 16);
  v16bf r;
#pragma unroll
  for (int i = 0; i < 8; ++i) { r[i] = lo[i]; r[i + 8] = hi[i]; }
  return r;
}

// ---------------- WMMA GEMM:  C[M,N] = A[M,K] * W[N,K]^T  (bf16 in, f32 out) ----------------
// One wave computes one 16x16 tile. K handled in chunks of 32.
__global__ __launch_bounds__(256) void wmma_gemm_bf16(
    const bf16_t* __restrict__ A, int lda,
    const bf16_t* __restrict__ W, int ldw,
    float* __restrict__ C, int ldc,
    int Mtiles, int Ntiles, int Kchunks) {
  const int lane = threadIdx.x & 31;
  const int tile = blockIdx.x * 8 + (threadIdx.x >> 5);
  const int tot  = Mtiles * Ntiles;
  if (tile >= tot) return;                 // wave-uniform: EXEC stays all-ones
  const int tn = tile % Ntiles;
  const int tm = tile / Ntiles;
  const int r    = lane & 15;
  const int half = lane >> 4;

  const bf16_t* ap = A + (size_t)(tm * 16 + r) * lda + half * 8;
  const bf16_t* wp = W + (size_t)(tn * 16 + r) * ldw + half * 8;

  v8f acc = {};
  for (int k = 0; k < Kchunks; ++k) {
    __builtin_prefetch(wp + 64, 0, 1);     // global_prefetch_b8 on weight stream
    v16bf af = load_frag(ap);  ap += 32;
    v16bf bf = load_frag(wp);  wp += 32;
    acc = __builtin_amdgcn_wmma_f32_16x16x32_bf16(
        /*neg_a=*/false, af, /*neg_b=*/false, bf,
        /*c_mod=*/(short)0, acc, /*reuse_a=*/false, /*reuse_b=*/false);
  }

  // C layout: lane holds (m = 8*half + v, n = r)
  float* cp = C + (size_t)(tm * 16 + 8 * half) * ldc + tn * 16 + r;
#pragma unroll
  for (int v = 0; v < 8; ++v) cp[(size_t)v * ldc] = acc[v];
}

// ---------------- f32 -> bf16 convert with zero padding ----------------
__global__ __launch_bounds__(256) void cvt_pad(
    const float* __restrict__ src, int srcStride, int srcRows, int srcCols,
    bf16_t* __restrict__ dst, int dstStride, int dstRows, int dstCols) {
  int idx = blockIdx.x * blockDim.x + threadIdx.x;
  int total = dstRows * dstCols;
  if (idx >= total) return;
  int rr = idx / dstCols, cc = idx % dstCols;
  float v = (rr < srcRows && cc < srcCols) ? src[(size_t)rr * srcStride + cc] : 0.0f;
  dst[(size_t)rr * dstStride + cc] = (bf16_t)v;
}

__global__ __launch_bounds__(256) void zero_bf16(bf16_t* __restrict__ p, int n) {
  int i = blockIdx.x * blockDim.x + threadIdx.x;
  if (i < n) p[i] = (bf16_t)0.0f;
}
__global__ __launch_bounds__(256) void zero_f32(float* __restrict__ p, int n) {
  int i = blockIdx.x * blockDim.x + threadIdx.x;
  if (i < n) p[i] = 0.0f;
}

// ---------------- GRU gate elementwise (torch gate order r,z,n) ----------------
// Gx includes x@Wih^T ; Gh includes h@Whh^T (biases added here, kept separate for gate n).
__global__ __launch_bounds__(256) void gru_gate(
    const float* __restrict__ Gx, const float* __restrict__ Gh, int ldg,
    const float* __restrict__ bih, const float* __restrict__ bhh,
    float* __restrict__ h,
    bf16_t* __restrict__ actA, int strideA,
    bf16_t* __restrict__ actB, int strideB) {
  int idx = blockIdx.x * blockDim.x + threadIdx.x;
  if (idx >= B * H) return;
  int b = idx / H, j = idx % H;
  const float* gx = Gx + (size_t)b * ldg;
  const float* gh = Gh + (size_t)b * ldg;
  float r = sigmoidf(gx[j] + bih[j] + gh[j] + bhh[j]);
  float z = sigmoidf(gx[H + j] + bih[H + j] + gh[H + j] + bhh[H + j]);
  float n = tanhf(gx[2 * H + j] + bih[2 * H + j] + r * (gh[2 * H + j] + bhh[2 * H + j]));
  float hp = h[idx];
  float hn = (1.0f - z) * n + z * hp;
  h[idx] = hn;
  bf16_t hb = (bf16_t)hn;
  actA[(size_t)b * strideA + j] = hb;
  if (actB) actB[(size_t)b * strideB + j] = hb;
}

// ---------------- planar-flow constants: u_hat and w.u_hat (data-independent) ----------------
__global__ void uhat_kernel(const float* __restrict__ fu, const float* __restrict__ fw,
                            float* __restrict__ uh, float* __restrict__ wdu) {
  int k = threadIdx.x;
  if (k >= L) return;
  float wu = 0.0f, ww = 0.0f;
  for (int j = 0; j < ZD; ++j) {
    wu += fw[k * ZD + j] * fu[k * ZD + j];
    ww += fw[k * ZD + j] * fw[k * ZD + j];
  }
  float m = -1.0f + softplusf(wu);
  float c = (m - wu) / (ww + 1e-6f);
  float d = 0.0f;
  for (int j = 0; j < ZD; ++j) {
    float u = fu[k * ZD + j] + c * fw[k * ZD + j];
    uh[k * ZD + j] = u;
    d += fw[k * ZD + j] * u;
  }
  wdu[k] = d;
}

// ---------------- latent sample + planar flows + log-det (1 thread / batch row) ----------------
__global__ __launch_bounds__(256) void zstep_kernel(
    const float* __restrict__ Gq,             // [B, NQ]  cols 0..15 -> Wqm, 16..31 -> Wqs
    const float* __restrict__ bqm, const float* __restrict__ bqs,
    const float* __restrict__ eps,            // [B,T,ZD]
    const float* __restrict__ fw, const float* __restrict__ fb,
    const float* __restrict__ uh, const float* __restrict__ wdu,
    float* __restrict__ out_z, float* __restrict__ out_mu, float* __restrict__ out_std,
    float* __restrict__ out_ld, float* __restrict__ ld_ws,
    bf16_t* __restrict__ actz, int actzStride, // Act_z z_prev slot (col offset pre-applied)
    bf16_t* __restrict__ zpad,                 // [B,T,KZ] bf16, p-GRU input
    int t) {
  int b = blockIdx.x * blockDim.x + threadIdx.x;
  if (b >= B) return;
  float zk[ZD], mu_[ZD], st_[ZD];
  const float* g  = Gq + (size_t)b * NQ;
  const float* ep = eps + ((size_t)b * T + t) * ZD;
#pragma unroll
  for (int j = 0; j < ZD; ++j) {
    float mu = g[j] + bqm[j];
    float s  = softplusf(g[ZD + j] + bqs[j]) + 1e-4f;
    mu_[j] = mu; st_[j] = s;
    zk[j] = mu + ep[j] * s;
  }
  float ld = (t == 0) ? 0.0f : ld_ws[b];
#pragma unroll
  for (int k = 0; k < L; ++k) {
    float a = fb[k];
#pragma unroll
    for (int j = 0; j < ZD; ++j) a += zk[j] * fw[k * ZD + j];
    float hh = tanhf(a);
#pragma unroll
    for (int j = 0; j < ZD; ++j) zk[j] += uh[k * ZD + j] * hh;
    float det = 1.0f + (1.0f - hh * hh) * wdu[k];
    ld += logf(fabsf(det) + 1e-6f);
  }
  ld_ws[b] = ld;
  if (t == T - 1) out_ld[b] = ld;

  size_t o = ((size_t)b * T + t) * ZD;
  bf16_t* zp = zpad + ((size_t)b * T + t) * KZ;
#pragma unroll
  for (int j = 0; j < ZD; ++j) {
    out_z[o + j]  = zk[j];
    out_mu[o + j] = mu_[j];
    out_std[o + j] = st_[j];
    actz[(size_t)b * actzStride + j] = (bf16_t)zk[j];
    zp[j]      = (bf16_t)zk[j];
    zp[ZD + j] = (bf16_t)0.0f;   // keep K-padding deterministic
  }
}

// ---------------- output projection elementwise ----------------
__global__ __launch_bounds__(256) void pout_kernel(
    const float* __restrict__ Gp,             // [B, NP] cols 0..37 Wpm, 38..75 Wps
    const float* __restrict__ bpm, const float* __restrict__ bps,
    float* __restrict__ recon_mu, float* __restrict__ recon_std, int t) {
  int idx = blockIdx.x * blockDim.x + threadIdx.x;
  if (idx >= B * XD) return;
  int b = idx / XD, d = idx % XD;
  const float* g = Gp + (size_t)b * NP;
  float m = g[d] + bpm[d];
  float s = softplusf(g[XD + d] + bps[d]) + 1e-4f;
  size_t o = ((size_t)b * T + t) * XD + d;
  recon_mu[o] = m;
  recon_std[o] = s;
}

// ---------------- host orchestration ----------------
static inline int cdiv(int a, int b) { return (a + b - 1) / b; }

extern "C" void kernel_launch(void* const* d_in, const int* in_sizes, int n_in,
                              void* d_out, int out_size, void* d_ws, size_t ws_size,
                              hipStream_t stream) {
  (void)in_sizes; (void)n_in; (void)out_size; (void)ws_size;
  const float* x     = (const float*)d_in[0];
  const float* eps   = (const float*)d_in[1];
  const float* Wih_q = (const float*)d_in[2];
  const float* Whh_q = (const float*)d_in[3];
  const float* bih_q = (const float*)d_in[4];
  const float* bhh_q = (const float*)d_in[5];
  const float* Wqm   = (const float*)d_in[6];
  const float* bqm   = (const float*)d_in[7];
  const float* Wqs   = (const float*)d_in[8];
  const float* bqs   = (const float*)d_in[9];
  const float* Wih_p = (const float*)d_in[10];
  const float* Whh_p = (const float*)d_in[11];
  const float* bih_p = (const float*)d_in[12];
  const float* bhh_p = (const float*)d_in[13];
  const float* Wpm   = (const float*)d_in[14];
  const float* bpm   = (const float*)d_in[15];
  const float* Wps   = (const float*)d_in[16];
  const float* bps   = (const float*)d_in[17];
  const float* fu    = (const float*)d_in[18];
  const float* fw    = (const float*)d_in[19];
  const float* fb    = (const float*)d_in[20];

  float* out       = (float*)d_out;
  float* recon_mu  = out;
  float* recon_std = recon_mu + (size_t)B * T * XD;
  float* out_z     = recon_std + (size_t)B * T * XD;
  float* out_mu    = out_z + (size_t)B * T * ZD;
  float* out_std   = out_mu + (size_t)B * T * ZD;
  float* out_ld    = out_std + (size_t)B * T * ZD;

  // workspace carve-out (256B aligned)
  char* base = (char*)d_ws;
  size_t off = 0;
  auto alloc = [&](size_t bytes) -> void* {
    off = (off + 255) & ~(size_t)255;
    void* p = base + off;
    off += bytes;
    return p;
  };
  bf16_t* Wihq_b = (bf16_t*)alloc((size_t)G3P * KX * 2);
  bf16_t* Whhq_b = (bf16_t*)alloc((size_t)G3P * KH * 2);
  bf16_t* Wihp_b = (bf16_t*)alloc((size_t)G3P * KZ * 2);
  bf16_t* Whhp_b = (bf16_t*)alloc((size_t)G3P * KH * 2);
  bf16_t* Wq_b   = (bf16_t*)alloc((size_t)NQ * KC * 2);
  bf16_t* Wp_b   = (bf16_t*)alloc((size_t)NP * KH * 2);
  bf16_t* Actx_q = (bf16_t*)alloc((size_t)B * KX * 2);
  bf16_t* Acth_q = (bf16_t*)alloc((size_t)B * KH * 2);
  bf16_t* Act_z  = (bf16_t*)alloc((size_t)B * KC * 2);
  bf16_t* Acth_p = (bf16_t*)alloc((size_t)B * KH * 2);
  bf16_t* zpad   = (bf16_t*)alloc((size_t)B * T * KZ * 2);
  float*  Gx_q   = (float*)alloc((size_t)B * G3P * 4);
  float*  Gh_q   = (float*)alloc((size_t)B * G3P * 4);
  float*  Gq     = (float*)alloc((size_t)B * NQ * 4);
  float*  Gx_p   = (float*)alloc((size_t)B * G3P * 4);
  float*  Gh_p   = (float*)alloc((size_t)B * G3P * 4);
  float*  Gp     = (float*)alloc((size_t)B * NP * 4);
  float*  h_q    = (float*)alloc((size_t)B * H * 4);
  float*  h_p    = (float*)alloc((size_t)B * H * 4);
  float*  ld_ws  = (float*)alloc((size_t)B * 4);
  float*  uh     = (float*)alloc((size_t)L * ZD * 4);
  float*  wdu    = (float*)alloc((size_t)L * 4);

  const int TPB = 256;
  auto cvt = [&](const float* src, int ss, int sr, int sc,
                 bf16_t* dst, int ds, int dr, int dc) {
    int tot = dr * dc;
    cvt_pad<<<cdiv(tot, TPB), TPB, 0, stream>>>(src, ss, sr, sc, dst, ds, dr, dc);
  };

  // ---- one-time (per call) setup ----
  cvt(Wih_q, XD, G3, XD, Wihq_b, KX, G3P, KX);
  cvt(Whh_q, H,  G3, H,  Whhq_b, KH, G3P, KH);
  cvt(Wih_p, ZD, G3, ZD, Wihp_b, KZ, G3P, KZ);
  cvt(Whh_p, H,  G3, H,  Whhp_b, KH, G3P, KH);
  cvt(Wqm, H + ZD, ZD, H + ZD, Wq_b,            KC, ZD, KC);
  cvt(Wqs, H + ZD, ZD, H + ZD, Wq_b + ZD * KC,  KC, ZD, KC);
  cvt(Wpm, H, XD, H, Wp_b,            KH, XD,      KH);
  cvt(Wps, H, XD, H, Wp_b + XD * KH,  KH, NP - XD, KH);   // rows 76..79 zero-filled

  zero_bf16<<<cdiv(B * KH, TPB), TPB, 0, stream>>>(Acth_q, B * KH);
  zero_bf16<<<cdiv(B * KC, TPB), TPB, 0, stream>>>(Act_z,  B * KC);
  zero_bf16<<<cdiv(B * KH, TPB), TPB, 0, stream>>>(Acth_p, B * KH);
  zero_f32 <<<cdiv(B * H,  TPB), TPB, 0, stream>>>(h_q, B * H);
  zero_f32 <<<cdiv(B * H,  TPB), TPB, 0, stream>>>(h_p, B * H);
  uhat_kernel<<<1, 32, 0, stream>>>(fu, fw, uh, wdu);

  auto gemm = [&](const bf16_t* A, int lda, const bf16_t* W, int ldw,
                  float* C, int ldc, int Mt, int Nt, int Kc_) {
    int tiles = Mt * Nt;
    wmma_gemm_bf16<<<cdiv(tiles, 8), TPB, 0, stream>>>(A, lda, W, ldw, C, ldc, Mt, Nt, Kc_);
  };
  const int MT = B / 16;        // 64
  const int NTG = G3P / 16;     // 94

  // ---- sequential time loop ----
  for (int t = 0; t < T; ++t) {
    // q-GRU
    cvt(x + (size_t)t * XD, T * XD, B, XD, Actx_q, KX, B, KX);
    gemm(Actx_q, KX, Wihq_b, KX, Gx_q, G3P, MT, NTG, KX / 32);
    gemm(Acth_q, KH, Whhq_b, KH, Gh_q, G3P, MT, NTG, KH / 32);
    gru_gate<<<cdiv(B * H, TPB), TPB, 0, stream>>>(
        Gx_q, Gh_q, G3P, bih_q, bhh_q, h_q, Acth_q, KH, Act_z, KC);

    // latent: [h, z_prev] @ [Wqm;Wqs]^T , then flows
    gemm(Act_z, KC, Wq_b, KC, Gq, NQ, MT, NQ / 16, KC / 32);
    zstep_kernel<<<cdiv(B, TPB), TPB, 0, stream>>>(
        Gq, bqm, bqs, eps, fw, fb, uh, wdu,
        out_z, out_mu, out_std, out_ld, ld_ws,
        Act_z + H, KC, zpad, t);

    // p-GRU (input = z_t just produced)
    gemm(zpad + (size_t)t * KZ, T * KZ, Wihp_b, KZ, Gx_p, G3P, MT, NTG, KZ / 32);
    gemm(Acth_p, KH, Whhp_b, KH, Gh_p, G3P, MT, NTG, KH / 32);
    gru_gate<<<cdiv(B * H, TPB), TPB, 0, stream>>>(
        Gx_p, Gh_p, G3P, bih_p, bhh_p, h_p, Acth_p, KH, (bf16_t*)nullptr, 0);

    // output projections
    gemm(Acth_p, KH, Wp_b, KH, Gp, NP, MT, NP / 16, KH / 32);
    pout_kernel<<<cdiv(B * XD, TPB), TPB, 0, stream>>>(Gp, bpm, bps, recon_mu, recon_std, t);
  }
}
